// CosineSALAD1_18846316495044
// MI455X (gfx1250) — compile-verified, matrix operand order checked by reading, not run on hardware
//
#include <hip/hip_runtime.h>
#include <math.h>

// ---------------- types ----------------
typedef __bf16  bf16_t;
typedef __attribute__((ext_vector_type(16))) __bf16 v16bf;
typedef __attribute__((ext_vector_type(8)))  float  v8f;
typedef __attribute__((ext_vector_type(4)))  unsigned int u32x4;
typedef __attribute__((ext_vector_type(8)))  int  i32x8;
typedef __attribute__((ext_vector_type(4)))  int  i32x4;

#define B_BATCH   64
#define N_ALL     2305
#define N_TOK     2304      // N-1
#define D_CH      256       // NUM_CHANNELS
#define C_DIM     128       // CLUSTER_DIM
#define K_CL      64        // NUM_CLUSTERS
#define T_DIM     256       // TOKEN_DIM
#define INV_REG   10.0f     // 1/REG
#define N_ITERS   5
#define OUT_DIM   (T_DIM + K_CL * C_DIM)   // 8448

#if defined(__has_builtin)
#if __has_builtin(__builtin_amdgcn_tensor_load_to_lds) && __has_builtin(__builtin_amdgcn_s_wait_tensorcnt)
#define HAVE_TDM 1
#endif
#endif
#ifndef HAVE_TDM
#define HAVE_TDM 0
#endif

// ---------------- WMMA helpers ----------------
__device__ __forceinline__ v8f wmma_bf16(v16bf a, v16bf b, v8f c) {
  // D = A(16x32 bf16) x B(32x16 bf16) + C(16x16 f32)
  return __builtin_amdgcn_wmma_f32_16x16x32_bf16(
      /*neg_a=*/false, a, /*neg_b=*/false, b,
      /*c_mod=*/(short)0, c, /*reuse_a=*/false, /*reuse_b=*/false);
}

// Load a 16x32 bf16 fragment (A, or B given row-major B^T) from row-major
// storage. p must point at (row = lane&15)'s data at k-offset 8*(lane>>4):
// halfs 0..7 from p[0..7], halfs 8..15 from p[16..23]. 16B aligned.
__device__ __forceinline__ v16bf frag_load(const bf16_t* p) {
  union { uint4 u[2]; v16bf v; } c;
  c.u[0] = *(const uint4*)(p);
  c.u[1] = *(const uint4*)(p + 16);
  return c.v;
}

// ---------------- K0a: normalize score anchors -> bf16 ----------------
__global__ __launch_bounds__(256) void k0_anchors(const float* __restrict__ sa,
                                                  bf16_t* __restrict__ saB) {
  __shared__ float red[256];
  int r = blockIdx.x, tid = threadIdx.x;
  float v = sa[r * D_CH + tid];
  red[tid] = v * v;
  __syncthreads();
  for (int off = 128; off > 0; off >>= 1) {
    if (tid < off) red[tid] += red[tid + off];
    __syncthreads();
  }
  float inv = 1.0f / fmaxf(sqrtf(red[0]), 1e-12f);
  saB[r * D_CH + tid] = (bf16_t)(v * inv);
}

// ---------------- K0b: convert f_proj_w -> bf16 ----------------
__global__ __launch_bounds__(256) void k0_wf(const float* __restrict__ wf,
                                             bf16_t* __restrict__ wfB) {
  int i = blockIdx.x * 256 + threadIdx.x;
  if (i < C_DIM * D_CH) wfB[i] = (bf16_t)wf[i];
}

// ---------------- Kt: token projection t = x[:,0] @ Wt^T + bt -------------
__global__ __launch_bounds__(256) void k_token(const float* __restrict__ x,
                                               const float* __restrict__ wt,
                                               const float* __restrict__ bt,
                                               float* __restrict__ tG) {
  int b = blockIdx.x, c = threadIdx.x;
  const float* xr = x + (size_t)b * N_ALL * D_CH;  // row 0
  const float* wr = wt + (size_t)c * D_CH;
  float s = 0.0f;
  #pragma unroll 4
  for (int d = 0; d < D_CH; ++d) s += xr[d] * wr[d];
  tG[b * T_DIM + c] = s + bt[c];
}

// ---------------- K1: fused feature projection + cosine scores -----------
// grid (36, 64), block 128 (4 waves). Tile = 64 rows of fi.
#define TILE_N 64
#define LDA    264   // 256 + 8 halfs padding (16B) -> bank-staggered rows
__global__ __launch_bounds__(128) void k1_proj_score(
    const float*  __restrict__ x,
    const bf16_t* __restrict__ wfB,    // (128,256) bf16 row-major
    const float*  __restrict__ fbias,  // (128)
    const bf16_t* __restrict__ saB,    // (64,256) bf16 row-major, normalized
    const float*  __restrict__ lsharp, // scalar
    const float*  __restrict__ dust,   // scalar
    bf16_t* __restrict__ fB,           // (B,2304,128) bf16
    float*  __restrict__ saug)         // (B,65,2304) f32
{
  __shared__ bf16_t sfi[TILE_N][LDA];
  __shared__ float  srn[TILE_N];
  int b   = blockIdx.y;
  int t0  = blockIdx.x * TILE_N;
  int tid = threadIdx.x;

  const float* xb = x + ((size_t)b * N_ALL + 1 + t0) * D_CH;
  // coalesced load + f32->bf16 convert into LDS
  for (int idx = tid; idx < TILE_N * D_CH; idx += 128) {
    int r = idx >> 8, c = idx & 255;
    sfi[r][c] = (bf16_t)xb[(size_t)r * D_CH + c];
  }
  __syncthreads();
  // per-row scale = exp(log_sharpness) / max(||fi||, eps)
  if (tid < TILE_N) {
    float ss = 0.0f;
    #pragma unroll 4
    for (int c = 0; c < D_CH; ++c) { float v = (float)sfi[tid][c]; ss += v * v; }
    srn[tid] = __expf(lsharp[0]) / fmaxf(sqrtf(ss), 1e-12f);
  }
  __syncthreads();

  int wave = tid >> 5, lane = tid & 31;
  int hi = lane >> 4, lo = lane & 15;

  v8f accF[8], accS[4];
  v8f z = {};
  #pragma unroll
  for (int i = 0; i < 8; ++i) accF[i] = z;
  #pragma unroll
  for (int i = 0; i < 4; ++i) accS[i] = z;

  #pragma unroll
  for (int kt = 0; kt < 8; ++kt) {
    int k0 = kt * 32;
    v16bf a = frag_load(&sfi[wave * 16 + lo][k0 + hi * 8]);
    #pragma unroll
    for (int ct = 0; ct < 8; ++ct) {  // f = fi @ Wf^T : B^T row = Wf row
      v16bf bb = frag_load(wfB + (size_t)(ct * 16 + lo) * D_CH + k0 + hi * 8);
      accF[ct] = wmma_bf16(a, bb, accF[ct]);
    }
    #pragma unroll
    for (int ct = 0; ct < 4; ++ct) {  // raw scores = fi @ sa_norm^T
      v16bf bb = frag_load(saB + (size_t)(ct * 16 + lo) * D_CH + k0 + hi * 8);
      accS[ct] = wmma_bf16(a, bb, accS[ct]);
    }
  }

  // epilogue: f (+bias) -> bf16, row-major (n, c)
  #pragma unroll
  for (int ct = 0; ct < 8; ++ct) {
    int c = ct * 16 + lo;
    float bias = fbias[c];
    #pragma unroll
    for (int r = 0; r < 8; ++r) {
      int row = wave * 16 + r + 8 * hi;
      fB[((size_t)b * N_TOK + t0 + row) * C_DIM + c] = (bf16_t)(accF[ct][r] + bias);
    }
  }
  // epilogue: s_aug[b][k][n] = raw * sharp/||fi||  (transposed store)
  #pragma unroll
  for (int ct = 0; ct < 4; ++ct) {
    int kcl = ct * 16 + lo;
    #pragma unroll
    for (int r = 0; r < 8; ++r) {
      int row = wave * 16 + r + 8 * hi;
      saug[((size_t)b * (K_CL + 1) + kcl) * N_TOK + t0 + row] = accS[ct][r] * srn[row];
    }
  }
  if (tid < TILE_N)
    saug[((size_t)b * (K_CL + 1) + K_CL) * N_TOK + t0 + tid] = dust[0];
}

// ---------------- K2: log-space Sinkhorn (5 iters), one block per batch ---
__global__ __launch_bounds__(1024) void k2_sinkhorn(const float* __restrict__ saug,
                                                    float* __restrict__ uG,
                                                    float* __restrict__ vG) {
  __shared__ float sv[N_TOK];
  __shared__ float su[K_CL + 1];
  int b = blockIdx.x, tid = threadIdx.x;
  int wave = tid >> 5, lane = tid & 31;
  const float* M = saug + (size_t)b * (K_CL + 1) * N_TOK;
  const float log_a = -logf((float)(K_CL + 1));
  const float log_b = -logf((float)N_TOK);

  for (int n = tid; n < N_TOK; n += 1024) sv[n] = 0.0f;
  __syncthreads();

  for (int it = 0; it < N_ITERS; ++it) {
    // u-pass: rows striped over waves; wave32 online logsumexp
    for (int k = wave; k < K_CL + 1; k += 32) {
      const float* row = M + (size_t)k * N_TOK;
      float m = -3.0e38f, s = 0.0f;
      for (int n = lane; n < N_TOK; n += 32) {
        float val = row[n] * INV_REG + sv[n];
        if (val > m) { s = s * __expf(m - val) + 1.0f; m = val; }
        else         { s += __expf(val - m); }
      }
      #pragma unroll
      for (int off = 16; off > 0; off >>= 1) {
        float om = __shfl_xor(m, off);
        float os = __shfl_xor(s, off);
        float nm = fmaxf(m, om);
        s = s * __expf(m - nm) + os * __expf(om - nm);
        m = nm;
      }
      if (lane == 0) su[k] = log_a - (m + __logf(s));
    }
    __syncthreads();
    // v-pass: columns striped over threads (coalesced per k)
    for (int n = tid; n < N_TOK; n += 1024) {
      float m = -3.0e38f, s = 0.0f;
      for (int k = 0; k < K_CL + 1; ++k) {
        float val = M[(size_t)k * N_TOK + n] * INV_REG + su[k];
        if (val > m) { s = s * __expf(m - val) + 1.0f; m = val; }
        else         { s += __expf(val - m); }
      }
      sv[n] = log_b - (m + __logf(s));
    }
    __syncthreads();
  }
  if (tid < K_CL + 1) uG[b * (K_CL + 1) + tid] = su[tid];
  for (int n = tid; n < N_TOK; n += 1024) vG[b * N_TOK + n] = sv[n];
}

// ---------------- TDM: DMA one 32x128 bf16 f-chunk into padded LDS -------
// D# per CDNA5 ISA ch.8: group0 = {count, lds_addr, global_addr, type=2};
// group1 = {data_size=2B, pad every 64 DWORDs by 4 DWORDs (-> 136-half rows),
//           tensor 128 x 2304, tile 128 x 32, dim0 stride 128}.
#define LDF 136   // 128 + 8 halfs padding
#if HAVE_TDM
__device__ __forceinline__ void tdm_load_f_chunk(const bf16_t* gsrc, unsigned lds_off) {
  unsigned long long ga = (unsigned long long)(uintptr_t)gsrc;
  u32x4 g0;
  g0.x = 1u;                                   // count=1 (valid user D#)
  g0.y = lds_off;                              // LDS byte address
  g0.z = (unsigned)(ga & 0xffffffffu);         // global_addr[31:0]
  g0.w = (unsigned)((ga >> 32) & 0x01ffffffu)  // global_addr[56:32]
         | (2u << 30);                         // type = 2 ("image")
  i32x8 g1;
  g1[0] = (int)((1u << 16)      // data_size = 1 -> 2 bytes
              | (1u << 20)      // pad_enable
              | (5u << 22)      // pad_interval: 2^(5+1)=64 DWORDs (one 128-elem row)
              | (3u << 25));    // pad_amount: 4 DWORDs (8 halfs)
  g1[1] = (int)(128u  << 16);   // tensor_dim0 = 128 (bits 79:48, low half)
  g1[2] = (int)(2304u << 16);   // tensor_dim1 = 2304 (bits 111:80, low half)
  g1[3] = (int)(128u  << 16);   // tile_dim0 = 128 (bits 127:112)
  g1[4] = 32;                   // tile_dim1 = 32
  g1[5] = 128;                  // tensor_dim0_stride = 128 (48b, low 32)
  g1[6] = 0;
  g1[7] = 0;
  i32x4 zz4 = {0, 0, 0, 0};     // groups 2/3 unused (2-D tile)
  i32x8 zz8 = {0, 0, 0, 0, 0, 0, 0, 0};  // extra group (6-arg toolchain form)
  __builtin_amdgcn_tensor_load_to_lds(g0, g1, zz4, zz4, zz8, 0);
}
#endif

// ---------------- K3: v_agg = exp(M+u+v) @ f - psum * anchors -------------
// grid 64 (batch), block 128 (4 waves): output 64x128 per batch.
// f chunks arrive via double-buffered TDM DMA (wave 0 drives the engine).
#define N_CHUNK (N_TOK / 32)   // 72
__global__ __launch_bounds__(128) void k3_agg(const float* __restrict__ saug,
                                              const float* __restrict__ uG,
                                              const float* __restrict__ vG,
                                              const bf16_t* __restrict__ fB,
                                              const float* __restrict__ agg,
                                              float* __restrict__ vagg) {
  __shared__ bf16_t sF[2][32][LDF];   // first LDS object -> offset 0
  __shared__ float  spsum[K_CL];
  int b = blockIdx.x, tid = threadIdx.x;
  int wave = tid >> 5, lane = tid & 31;
  int hi = lane >> 4, lo = lane & 15;

  int k = wave * 16 + lo;                  // A-fragment row (cluster index)
  float uk = uG[b * (K_CL + 1) + k];
  const float* Mrow = saug + ((size_t)b * (K_CL + 1) + k) * N_TOK;
  const float* vb = vG + (size_t)b * N_TOK;
  const bf16_t* fb = fB + (size_t)b * N_TOK * C_DIM;

  v8f acc[8];
  v8f z = {};
  #pragma unroll
  for (int i = 0; i < 8; ++i) acc[i] = z;
  float psum = 0.0f;

#if HAVE_TDM
  unsigned sf_lds[2];
  sf_lds[0] = (unsigned)(uintptr_t)&sF[0][0][0];  // LDS offset = addr[31:0]
  sf_lds[1] = (unsigned)(uintptr_t)&sF[1][0][0];
  if (wave == 0) tdm_load_f_chunk(fb, sf_lds[0]); // prime chunk 0
#endif

  for (int ci = 0; ci < N_CHUNK; ++ci) {
    int n0 = ci * 32;
    int buf = ci & 1;
#if HAVE_TDM
    if (wave == 0) __builtin_amdgcn_s_wait_tensorcnt((short)0);  // chunk ci landed
    __syncthreads();                                             // publish; prev buf free
    if (wave == 0 && ci + 1 < N_CHUNK)
      tdm_load_f_chunk(fb + (size_t)(n0 + 32) * C_DIM, sf_lds[buf ^ 1]);
#else
    __syncthreads();
    for (int idx = tid; idx < 32 * C_DIM; idx += 128) {
      int r = idx >> 7, c = idx & 127;
      sF[buf][r][c] = fb[(size_t)(n0 + r) * C_DIM + c];
    }
    __syncthreads();
#endif

    // A fragment = p = exp(M/reg + u + v), built in registers
    int nA = n0 + hi * 8;
    float mv[16], vv[16];
    *(float4*)&mv[0]  = *(const float4*)(Mrow + nA);
    *(float4*)&mv[4]  = *(const float4*)(Mrow + nA + 4);
    *(float4*)&mv[8]  = *(const float4*)(Mrow + nA + 16);
    *(float4*)&mv[12] = *(const float4*)(Mrow + nA + 20);
    *(float4*)&vv[0]  = *(const float4*)(vb + nA);
    *(float4*)&vv[4]  = *(const float4*)(vb + nA + 4);
    *(float4*)&vv[8]  = *(const float4*)(vb + nA + 16);
    *(float4*)&vv[12] = *(const float4*)(vb + nA + 20);
    v16bf a;
    #pragma unroll
    for (int j = 0; j < 16; ++j) {
      float p = __expf(mv[j] * INV_REG + uk + vv[j]);
      psum += p;
      a[j] = (bf16_t)p;
    }
    // B fragments from staged f chunk (B^T access via LDS)
    #pragma unroll
    for (int ct = 0; ct < 8; ++ct) {
      int c = ct * 16 + lo;
      v16bf bb;
      #pragma unroll
      for (int j = 0; j < 8; ++j) bb[j] = sF[buf][hi * 8 + j][c];
      #pragma unroll
      for (int j = 0; j < 8; ++j) bb[8 + j] = sF[buf][16 + hi * 8 + j][c];
      acc[ct] = wmma_bf16(a, bb, acc[ct]);
    }
  }

  // fold psum across the two lane halves holding the same A row
  float ps = psum + __shfl_xor(psum, 16);
  if (hi == 0) spsum[wave * 16 + lo] = ps;
  __syncthreads();

  #pragma unroll
  for (int ct = 0; ct < 8; ++ct) {
    int c = ct * 16 + lo;
    #pragma unroll
    for (int r = 0; r < 8; ++r) {
      int krow = wave * 16 + r + 8 * hi;
      float val = acc[ct][r] - spsum[krow] * agg[krow * C_DIM + c];
      vagg[((size_t)b * K_CL + krow) * C_DIM + c] = val;
    }
  }
}

// ---------------- K4: concat(t, v_agg) and row L2-normalize ---------------
__global__ __launch_bounds__(256) void k4_norm(const float* __restrict__ tG,
                                               const float* __restrict__ vagg,
                                               float* __restrict__ out) {
  __shared__ float red[256];
  int b = blockIdx.x, tid = threadIdx.x;
  const float* ta = tG + (size_t)b * T_DIM;
  const float* va = vagg + (size_t)b * K_CL * C_DIM;
  float ss = 0.0f;
  for (int idx = tid; idx < OUT_DIM; idx += 256) {
    float v = (idx < T_DIM) ? ta[idx] : va[idx - T_DIM];
    ss += v * v;
  }
  red[tid] = ss;
  __syncthreads();
  for (int off = 128; off > 0; off >>= 1) {
    if (tid < off) red[tid] += red[tid + off];
    __syncthreads();
  }
  float inv = 1.0f / fmaxf(sqrtf(red[0]), 1e-12f);
  float* ob = out + (size_t)b * OUT_DIM;
  for (int idx = tid; idx < OUT_DIM; idx += 256) {
    float v = (idx < T_DIM) ? ta[idx] : va[idx - T_DIM];
    ob[idx] = v * inv;
  }
}

// ---------------- launch ----------------
extern "C" void kernel_launch(void* const* d_in, const int* in_sizes, int n_in,
                              void* d_out, int out_size, void* d_ws, size_t ws_size,
                              hipStream_t stream) {
  const float* x      = (const float*)d_in[0];   // (64,2305,256)
  const float* wf     = (const float*)d_in[1];   // (128,256)
  const float* fbias  = (const float*)d_in[2];   // (128)
  const float* wt     = (const float*)d_in[3];   // (256,256)
  const float* bt     = (const float*)d_in[4];   // (256)
  const float* sa     = (const float*)d_in[5];   // (64,256)
  const float* agg    = (const float*)d_in[6];   // (64,128)
  const float* dust   = (const float*)d_in[7];   // scalar
  const float* lsharp = (const float*)d_in[8];   // scalar
  float* out = (float*)d_out;

  char* ws = (char*)d_ws;
  size_t off = 0;
  auto take = [&](size_t bytes) -> void* {
    void* p = ws + off;
    off += (bytes + 255) & ~(size_t)255;
    return p;
  };
  bf16_t* wfB  = (bf16_t*)take((size_t)C_DIM * D_CH * sizeof(bf16_t));
  bf16_t* saB  = (bf16_t*)take((size_t)K_CL * D_CH * sizeof(bf16_t));
  float*  tG   = (float*) take((size_t)B_BATCH * T_DIM * sizeof(float));
  bf16_t* fB   = (bf16_t*)take((size_t)B_BATCH * N_TOK * C_DIM * sizeof(bf16_t));
  float*  saug = (float*) take((size_t)B_BATCH * (K_CL + 1) * N_TOK * sizeof(float));
  float*  uG   = (float*) take((size_t)B_BATCH * (K_CL + 1) * sizeof(float));
  float*  vG   = (float*) take((size_t)B_BATCH * N_TOK * sizeof(float));
  float*  vagg = (float*) take((size_t)B_BATCH * K_CL * C_DIM * sizeof(float));
  (void)ws_size; (void)in_sizes; (void)n_in; (void)out_size;

  k0_anchors<<<K_CL, 256, 0, stream>>>(sa, saB);
  k0_wf<<<(C_DIM * D_CH + 255) / 256, 256, 0, stream>>>(wf, wfB);
  k_token<<<B_BATCH, 256, 0, stream>>>(x, wt, bt, tG);

  dim3 g1(N_TOK / TILE_N, B_BATCH);   // (36, 64)
  k1_proj_score<<<g1, 128, 0, stream>>>(x, wfB, fbias, saB, lsharp, dust, fB, saug);

  k2_sinkhorn<<<B_BATCH, 1024, 0, stream>>>(saug, uG, vG);

  k3_agg<<<B_BATCH, 128, 0, stream>>>(saug, uG, vG, fB, agg, vagg);

  k4_norm<<<B_BATCH, 256, 0, stream>>>(tG, vagg, out);
}
